// MDGCNBlock_57114475102443
// MI455X (gfx1250) — matrix operand order repaired and checked by reading, not compiled, not run on hardware
//
#include <hip/hip_runtime.h>
#include <math.h>

typedef unsigned short u16;
typedef __attribute__((ext_vector_type(16))) __bf16 v16bf;
typedef __attribute__((ext_vector_type(8)))  float  v8f;

union Frag { v16bf v; uint4 q[2]; };

// ---- WMMA fragment loaders (row-major bf16 source, 16-bit layouts per ISA 7.12.2) ----
__device__ __forceinline__ void load_a_frag(Frag& f, const u16* base, int stride, int row0, int kb, int lane){
  int r = lane & 15, half = lane >> 4;
  const u16* p = base + (size_t)(row0 + r) * stride + kb + 8*half;
  f.q[0] = *(const uint4*)p;          // K = kb+8h .. +7
  f.q[1] = *(const uint4*)(p + 16);   // K = kb+16+8h .. +7
}
__device__ __forceinline__ void load_b_frag(Frag& f, const u16* base, int stride, int col0, int kb, int lane){
  int n = lane & 15, half = lane >> 4;
  const u16* p = base + (size_t)(col0 + n) * stride + kb + 8*half;
  f.q[0] = *(const uint4*)p;          // K = kb+8h .. +7
  f.q[1] = *(const uint4*)(p + 16);   // K = kb+16+8h .. +7
}
// bf16x3: dot ~= hh + hl + lh (ll term ~2^-32, dropped)
__device__ __forceinline__ v8f mma_bf16x3(v8f c, const Frag& ah, const Frag& al, const Frag& bh, const Frag& bl){
  c = __builtin_amdgcn_wmma_f32_16x16x32_bf16(false, ah.v, false, bh.v, (short)0, c, false, false);
  c = __builtin_amdgcn_wmma_f32_16x16x32_bf16(false, ah.v, false, bl.v, (short)0, c, false, false);
  c = __builtin_amdgcn_wmma_f32_16x16x32_bf16(false, al.v, false, bh.v, (short)0, c, false, false);
  return c;
}
__device__ __forceinline__ void split_bf(float x, u16& h, u16& l){
  union { float f; unsigned u; } a; a.f = x;
  unsigned hu = a.u & 0xffff0000u;
  h = (u16)(hu >> 16);
  union { unsigned u; float f; } bb; bb.u = hu;
  union { float f; unsigned u; } cc; cc.f = x - bb.f;
  l = (u16)((cc.u + 0x8000u) >> 16);
}
// register-resident sorted top-9 insert (ascending; ties -> lower index, matching jax top_k)
__device__ __forceinline__ void top9_insert(float (&bv)[9], int (&bi)[9], float d, int gi){
  if (!((d < bv[8]) || (d == bv[8] && gi < bi[8]))) return;
  bool act = true;
#pragma unroll
  for (int k = 8; k > 0; --k){
    if (act){
      bool c = (d < bv[k-1]) || (d == bv[k-1] && gi < bi[k-1]);
      if (c){ bv[k]=bv[k-1]; bi[k]=bi[k-1]; }
      else  { bv[k]=d; bi[k]=gi; act=false; }
    }
  }
  if (act){ bv[0]=d; bi[0]=gi; }
}

// ---- prep: concat x|y -> feat0 fp32 + bf16 hi/lo points ----
__global__ void prep_points_kernel(const float* __restrict__ x, const float* __restrict__ y,
                                   float* __restrict__ feat0, u16* __restrict__ Ph, u16* __restrict__ Pl){
  size_t i = (size_t)blockIdx.x * 256 + threadIdx.x;           // B*8192*128
  if (i >= (size_t)4*8192*128) return;
  size_t b = i / ((size_t)8192*128); size_t r = i % ((size_t)8192*128);
  size_t n = r / 128, c = r % 128;
  float v = (n < 4096) ? x[(b*4096+n)*128 + c] : y[(b*4096 + (n-4096))*128 + c];
  feat0[i] = v;
  u16 h,l; split_bf(v,h,l); Ph[i]=h; Pl[i]=l;
}
__global__ void norms_kernel(const float* __restrict__ feat, float* __restrict__ norms){
  int p = blockIdx.x * 256 + threadIdx.x;                       // B*8192
  if (p >= 4*8192) return;
  const float4* f = (const float4*)(feat + (size_t)p*128);
  float s = 0.f;
#pragma unroll 8
  for (int i=0;i<32;++i){ float4 q = f[i]; s += q.x*q.x + q.y*q.y + q.z*q.z + q.w*q.w; }
  norms[p] = s;
}
__global__ void split_kernel(const float* __restrict__ src, u16* __restrict__ hi, u16* __restrict__ lo, int n){
  int i = blockIdx.x * 256 + threadIdx.x; if (i >= n) return;
  u16 h,l; split_bf(src[i],h,l); hi[i]=h; lo[i]=l;
}
__global__ void bnfold_kernel(const float* __restrict__ g, const float* __restrict__ beta,
                              const float* __restrict__ m, const float* __restrict__ v,
                              const float* __restrict__ bias, float* __restrict__ s, float* __restrict__ t, int n){
  int i = blockIdx.x * 256 + threadIdx.x; if (i >= n) return;
  float sc = g[i] * rsqrtf(v[i] + 1e-5f);
  s[i] = sc; t[i] = (bias[i] - m[i]) * sc + beta[i];
}

// ---- fused distance GEMM + top-9 KNN ----
#define LDSW 1028
__global__ __launch_bounds__(256) void knn_kernel(const u16* __restrict__ Ph, const u16* __restrict__ Pl,
                                                  const float* __restrict__ norms, int* __restrict__ nbr){
  __shared__ float tn[1024];
  __shared__ float sn[16];
  struct Cand { float v[2304]; int i[2304]; };
  union Smem { float dot[16*LDSW]; Cand cand; };
  __shared__ Smem sm;

  int bid = blockIdx.x;
  int tile = bid & 255; bid >>= 8;
  int tgt  = bid & 1;   bid >>= 1;
  int src  = bid & 1;   int b = bid >> 1;
  int lane = threadIdx.x & 31, wave = threadIdx.x >> 5;
  const size_t pbase = (size_t)b * 8192 * 128;
  const int srcRow0 = src*4096 + tile*16;
  const int tgtRow0 = tgt*4096;

  Frag ah[4], al[4];
#pragma unroll
  for (int ks=0; ks<4; ++ks){
    load_a_frag(ah[ks], Ph + pbase, 128, srcRow0, ks*32, lane);
    load_a_frag(al[ks], Pl + pbase, 128, srcRow0, ks*32, lane);
  }
  if (threadIdx.x < 16) sn[threadIdx.x] = norms[(size_t)b*8192 + srcRow0 + threadIdx.x];

  float bv[9]; int bi[9];
#pragma unroll
  for (int j=0;j<9;++j){ bv[j] = INFINITY; bi[j] = 0x7fffffff; }
  const int row = threadIdx.x >> 4, seg = threadIdx.x & 15;

  for (int chunk = 0; chunk < 4; ++chunk){
    __syncthreads();                                   // prior scan done before overwriting tn/dot
    for (int i = threadIdx.x; i < 1024; i += 256)
      tn[i] = norms[(size_t)b*8192 + tgtRow0 + chunk*1024 + i];
    for (int t = 0; t < 8; ++t){
      int ct = wave*8 + t;
      int col0 = tgtRow0 + chunk*1024 + ct*16;
      v8f acc = {};
#pragma unroll
      for (int ks=0; ks<4; ++ks){
        Frag bh, bl;
        load_b_frag(bh, Ph + pbase, 128, col0, ks*32, lane);
        load_b_frag(bl, Pl + pbase, 128, col0, ks*32, lane);
        acc = mma_bf16x3(acc, ah[ks], al[ks], bh, bl);
      }
      int n = lane & 15, half = lane >> 4;
#pragma unroll
      for (int v=0; v<8; ++v)
        sm.dot[(8*half+v)*LDSW + ct*16 + n] = -2.0f * acc[v];
    }
    __syncthreads();
    float rn = sn[row];
    for (int j=0; j<64; ++j){
      int c = seg + (j << 4);                          // interleaved -> conflict-free LDS reads
      float d = rn + tn[c] + sm.dot[row*LDSW + c];
      top9_insert(bv, bi, d, chunk*1024 + c);
    }
  }
  __syncthreads();
  {
    int base = threadIdx.x * 9;
#pragma unroll
    for (int j=0;j<9;++j){ sm.cand.v[base+j]=bv[j]; sm.cand.i[base+j]=bi[j]; }
  }
  __syncthreads();
  if (threadIdx.x < 16){
    int r = threadIdx.x;
    float mv[9]; int mi[9];
#pragma unroll
    for (int j=0;j<9;++j){ mv[j]=INFINITY; mi[j]=0x7fffffff; }
    for (int t=0;t<16;++t){
      int base = (r*16 + t)*9;
      for (int j=0;j<9;++j) top9_insert(mv, mi, sm.cand.v[base+j], sm.cand.i[base+j]);
    }
    int slot = (tgt == src) ? 0 : 9;                   // same-set neighbors first (xx / yy)
    size_t obase = ((size_t)b*8192 + srcRow0 + r)*18 + slot;
#pragma unroll
    for (int j=0;j<9;++j) nbr[obase + j] = tgt*4096 + mi[j];
  }
}

// ---- gather + max -> interleaved h=[f,msg] (bf16 hi/lo) ----
__global__ void gather_kernel(const float* __restrict__ feat, const int* __restrict__ nbr,
                              u16* __restrict__ Hh, u16* __restrict__ Hl){
  int p = blockIdx.x;            // B*8192
  int c = threadIdx.x;           // 128
  int b = p >> 13;
  float f = feat[(size_t)p*128 + c];
  const int* nb = nbr + (size_t)p*18;
  float mx = -INFINITY;
#pragma unroll
  for (int j=0;j<18;++j){
    int idx = nb[j];
    float v = feat[((size_t)(b << 13) + idx)*128 + c] - f;
    mx = fmaxf(mx, v);
  }
  size_t base = (size_t)p * 256;
  u16 h,l;
  split_bf(f, h, l);  Hh[base + 2*c]   = h; Hl[base + 2*c]   = l;   // stack(...,-1).reshape -> interleave
  split_bf(mx, h, l); Hh[base + 2*c+1] = h; Hl[base + 2*c+1] = l;
}

// ---- generic WMMA GEMM + BN + (GELU) + (residual) + (mask/final) ----
__global__ __launch_bounds__(256) void gemm_kernel(
    const u16* __restrict__ Ah, const u16* __restrict__ Al,
    const u16* __restrict__ Wh, const u16* __restrict__ Wl,
    int K, int Nout,
    const float* __restrict__ s_arr, const float* __restrict__ t_arr,
    int do_gelu, const float* __restrict__ residual,
    float* __restrict__ out_f32, u16* __restrict__ out_hi, u16* __restrict__ out_lo,
    float* __restrict__ final_out, const float* __restrict__ xmask, const float* __restrict__ ymask)
{
  int lane = threadIdx.x & 31, wave = threadIdx.x >> 5;
  int nt = blockIdx.y * 8 + wave;
  if (nt * 16 >= Nout) return;
  int row0 = blockIdx.x * 16, col0 = nt * 16;
  v8f acc = {};
  for (int kb = 0; kb < K; kb += 32){
    Frag ah, al, bh, bl;
    load_a_frag(ah, Ah, K, row0, kb, lane);
    load_a_frag(al, Al, K, row0, kb, lane);
    load_b_frag(bh, Wh, K, col0, kb, lane);   // W is [Nout][K] row-major: lane n reads its own W row
    load_b_frag(bl, Wl, K, col0, kb, lane);
    acc = mma_bf16x3(acc, ah, al, bh, bl);
  }
  int o = col0 + (lane & 15);
  int half = lane >> 4;
  float sc = s_arr[o], sh = t_arr[o];
#pragma unroll
  for (int v=0; v<8; ++v){
    int rowi = row0 + 8*half + v;
    size_t row = (size_t)rowi;
    float val = acc[v]*sc + sh;
    if (do_gelu) val = 0.5f*val*(1.0f + erff(val*0.7071067811865475f));
    if (residual) val += residual[row*128 + o];
    if (final_out){
      int bb = rowi >> 13, wi = rowi & 8191;
      if (wi < 4096)
        final_out[((size_t)(bb*4096 + wi))*128 + o] = val * xmask[bb*4096 + wi];
      else
        final_out[(size_t)4*4096*128 + ((size_t)(bb*4096 + (wi-4096)))*128 + o] = val * ymask[bb*4096 + (wi-4096)];
    }
    if (out_f32) out_f32[row*Nout + o] = val;
    if (out_hi){ u16 h,l; split_bf(val,h,l); out_hi[row*Nout + o]=h; out_lo[row*Nout + o]=l; }
  }
}

extern "C" void kernel_launch(void* const* d_in, const int* in_sizes, int n_in,
                              void* d_out, int out_size, void* d_ws, size_t ws_size,
                              hipStream_t stream) {
  (void)in_sizes; (void)n_in; (void)out_size; (void)ws_size;
  const float* x     = (const float*)d_in[0];
  const float* y     = (const float*)d_in[1];
  const float* xmask = (const float*)d_in[2];
  const float* ymask = (const float*)d_in[3];
  const float *Wt[5], *Wb[5], *Wg[5], *Wbe[5], *Wm[5], *Wv[5];
  for (int i=0;i<5;++i){
    Wt[i]  = (const float*)d_in[4+6*i+0];
    Wb[i]  = (const float*)d_in[4+6*i+1];
    Wg[i]  = (const float*)d_in[4+6*i+2];
    Wbe[i] = (const float*)d_in[4+6*i+3];
    Wm[i]  = (const float*)d_in[4+6*i+4];
    Wv[i]  = (const float*)d_in[4+6*i+5];
  }
  const int Wo[5] = {128,128,128,512,128};   // mr0, mr1, fc, f1, f2
  const int Wk[5] = {256,256,128,128,512};

  const int P = 4*8192;
  char* ws = (char*)d_ws;
  size_t off = 0;
  auto alloc = [&](size_t bytes)->char*{ char* p = ws + off; off = (off + bytes + 255) & ~(size_t)255; return p; };

  float* feat0  = (float*)alloc((size_t)P*128*4);
  float* feat1  = (float*)alloc((size_t)P*128*4);
  float* feat2  = (float*)alloc((size_t)P*128*4);
  float* fcout  = (float*)alloc((size_t)P*128*4);
  u16*   ptsH   = (u16*)  alloc((size_t)P*128*2);
  u16*   ptsL   = (u16*)  alloc((size_t)P*128*2);
  u16*   f2H    = (u16*)  alloc((size_t)P*128*2);
  u16*   f2L    = (u16*)  alloc((size_t)P*128*2);
  u16*   fcH    = (u16*)  alloc((size_t)P*128*2);
  u16*   fcL    = (u16*)  alloc((size_t)P*128*2);
  u16*   hH     = (u16*)  alloc((size_t)P*256*2);
  u16*   hL     = (u16*)  alloc((size_t)P*256*2);
  u16*   g1H    = (u16*)  alloc((size_t)P*512*2);
  u16*   g1L    = (u16*)  alloc((size_t)P*512*2);
  float* nrm    = (float*)alloc((size_t)P*4);
  int*   nbr    = (int*)  alloc((size_t)P*18*4);
  u16 *WH[5], *WL[5]; float *BS[5], *BT[5];
  for (int i=0;i<5;++i){
    WH[i] = (u16*)alloc((size_t)Wo[i]*Wk[i]*2);
    WL[i] = (u16*)alloc((size_t)Wo[i]*Wk[i]*2);
    BS[i] = (float*)alloc((size_t)Wo[i]*4);
    BT[i] = (float*)alloc((size_t)Wo[i]*4);
  }

  // prep
  prep_points_kernel<<<16384, 256, 0, stream>>>(x, y, feat0, ptsH, ptsL);
  norms_kernel<<<128, 256, 0, stream>>>(feat0, nrm);
  for (int i=0;i<5;++i){
    int nw = Wo[i]*Wk[i];
    split_kernel<<<(nw+255)/256, 256, 0, stream>>>(Wt[i], WH[i], WL[i], nw);
    bnfold_kernel<<<(Wo[i]+255)/256, 256, 0, stream>>>(Wg[i], Wbe[i], Wm[i], Wv[i], Wb[i], BS[i], BT[i], Wo[i]);
  }

  // KNN: 4 batches x {src x,y} x {tgt x,y} x 256 row tiles
  knn_kernel<<<4096, 256, 0, stream>>>(ptsH, ptsL, nrm, nbr);

  // mr0
  gather_kernel<<<P, 128, 0, stream>>>(feat0, nbr, hH, hL);
  gemm_kernel<<<dim3(P/16, 1), 256, 0, stream>>>(hH, hL, WH[0], WL[0], 256, 128, BS[0], BT[0],
      1, feat0, feat1, nullptr, nullptr, nullptr, nullptr, nullptr);
  // mr1 (also emit bf16 for fc input)
  gather_kernel<<<P, 128, 0, stream>>>(feat1, nbr, hH, hL);
  gemm_kernel<<<dim3(P/16, 1), 256, 0, stream>>>(hH, hL, WH[1], WL[1], 256, 128, BS[1], BT[1],
      1, feat1, feat2, f2H, f2L, nullptr, nullptr, nullptr);
  // fc (residual = ORIGINAL x|y = feat0), emit bf16 for f1 input
  gemm_kernel<<<dim3(P/16, 1), 256, 0, stream>>>(f2H, f2L, WH[2], WL[2], 128, 128, BS[2], BT[2],
      0, feat0, fcout, fcH, fcL, nullptr, nullptr, nullptr);
  // ffn f1 (no residual), bf16 out only
  gemm_kernel<<<dim3(P/16, 4), 256, 0, stream>>>(fcH, fcL, WH[3], WL[3], 128, 512, BS[3], BT[3],
      1, nullptr, nullptr, g1H, g1L, nullptr, nullptr, nullptr);
  // ffn f2: gelu + residual(fcout) + mask -> final split output
  gemm_kernel<<<dim3(P/16, 1), 256, 0, stream>>>(g1H, g1L, WH[4], WL[4], 512, 128, BS[4], BT[4],
      1, fcout, nullptr, nullptr, nullptr, (float*)d_out, xmask, ymask);
}
